// RNN_generator_28011776705060
// MI455X (gfx1250) — matrix operand order repaired and checked by reading, not compiled, not run on hardware
//
#include <hip/hip_runtime.h>
#include <hip/hip_bf16.h>

// ---------------------------------------------------------------------------
// Persistent GRU forward for MI455X (gfx1250, wave32, WMMA, 320KB-LDS WGP).
//   tokens [64,512] i32, emb [32000,256] f32, Wk [256,3072] f32,
//   Uk [1024,3072] f32, b [2,3072] f32, Wout [1024,32000] f32
//   out = hT @ Wout  -> [64, 32000] f32
// Design: ONE persistent kernel runs all 512 timesteps.  Each of the 64
// workgroups owns a 16-column tile of H, stages its step-invariant weight
// slice (Uk 96KB + Wk 24KB, bf16, bank-rotated rows) into LDS once via the
// Tensor Data Mover (TDM), and syncs with the other WGs via a global atomic
// barrier between steps.  Weights never touch L2 again after staging.
// ---------------------------------------------------------------------------

typedef __bf16 bf16_t;
typedef bf16_t v16bf __attribute__((ext_vector_type(16)));
typedef bf16_t v8bf  __attribute__((ext_vector_type(8)));
typedef float  v8f   __attribute__((ext_vector_type(8)));
typedef float  v2f   __attribute__((ext_vector_type(2)));
typedef unsigned u32x4 __attribute__((ext_vector_type(4)));
typedef int      i32x8 __attribute__((ext_vector_type(8)));
typedef int      i32x4 __attribute__((ext_vector_type(4)));

#define VOCAB 32000
#define EMBED 256
#define HID   1024
#define B3    3072
#define BATCH 64
#define SEQ   512
#define NWG   64          // persistent grid size (one WG per 16-col tile of H)

// LDS row strides (elements).  2064B row stride = 516 dwords == 4 (mod 64)
// banks -> lanes 0..15 land on distinct bank quads for ds_load_b128.
#define UK_ROWPAD 1032    // 1024 + 8
#define WK_ROWPAD 264     // 256 + 8
#define LDS_UK_ELEMS (3 * 16 * UK_ROWPAD)            // 49,536
#define LDS_WK_ELEMS (3 * 16 * WK_ROWPAD)            // 12,672
#define LDS_BYTES ((LDS_UK_ELEMS + LDS_WK_ELEMS) * 2) // 124,416 B

#if defined(__has_builtin)
#if __has_builtin(__builtin_amdgcn_tensor_load_to_lds) && \
    __has_builtin(__builtin_amdgcn_s_wait_tensorcnt)
#define HAVE_TDM 1
#endif
#endif
#ifndef HAVE_TDM
#define HAVE_TDM 0
#endif

#if HAVE_TDM
// 1D TDM tile: n8 x 8-byte elements, global -> LDS byte offset.
// D# group0: count=1, lds_addr, global_addr[56:0], type=2 ("image").
// D# group1: data_size=3 (8B), tensor_dim0=tile_dim0=n8, tensor_dim1=tile_dim1=1.
__device__ __forceinline__ void tdm_load_row(unsigned lds_byte_off,
                                             const void* gsrc, unsigned n8) {
  unsigned long long ga = (unsigned long long)gsrc;
  u32x4 g0;
  g0.x = 1u;                                                // count=1
  g0.y = lds_byte_off;                                      // lds_addr
  g0.z = (unsigned)(ga & 0xffffffffu);                      // global_addr lo
  g0.w = (unsigned)((ga >> 32) & 0x01ffffffu) | (2u << 30); // addr[56:32],type=2
  i32x8 g1;
  g1[0] = (int)(3u << 16);   // workgroup_mask=0, data_size=3 (8B), no flags
  g1[1] = (int)(n8 << 16);   // tensor_dim0[15:0] in bits[31:16]
  g1[2] = (int)(1u << 16);   // tensor_dim0 hi=0, tensor_dim1=1
  g1[3] = (int)(n8 << 16);   // tensor_dim1 hi=0, tile_dim0=n8
  g1[4] = 1;                 // tile_dim1=1, tile_dim2=0
  g1[5] = (int)n8;           // tensor_dim0_stride lo (unused for 1 row)
  g1[6] = 0;
  g1[7] = 0;
  i32x4 z4 = {0, 0, 0, 0};
#if __clang_major__ >= 23
  i32x8 z8 = {0, 0, 0, 0, 0, 0, 0, 0};
  __builtin_amdgcn_tensor_load_to_lds(g0, g1, z4, z4, z8, 0);
#else
  __builtin_amdgcn_tensor_load_to_lds(g0, g1, z4, z4, 0);
#endif
}
#endif  // HAVE_TDM

// A-fragment (16x32 bf16, MxK): lane=(half,m); elems 0..15 at k=16*half+0..15.
// Two contiguous 16B loads.
__device__ __forceinline__ v16bf ld_a_frag(const bf16_t* p) {
  v8bf lo = *(const v8bf*)(p);
  v8bf hi = *(const v8bf*)(p + 8);
  return __builtin_shufflevector(lo, hi, 0, 1, 2, 3, 4, 5, 6, 7,
                                         8, 9, 10, 11, 12, 13, 14, 15);
}

__device__ __forceinline__ v8f wmma_bf16(v16bf a, v16bf b, v8f c) {
  return __builtin_amdgcn_wmma_f32_16x16x32_bf16(
      false, a, false, b, (short)0, c, false, false);
}

__device__ __forceinline__ float sigm(float x) {
  return 1.0f / (1.0f + __expf(-x));
}
__device__ __forceinline__ float tanh_fast(float x) {
  float t = __expf(2.0f * x);
  return (t - 1.0f) / (t + 1.0f);
}

// ------------------------- one-time prep kernels ---------------------------

__global__ void gru_transpose_bf16(const float* __restrict__ src,
                                   bf16_t* __restrict__ dst, int R, int C) {
  size_t idx = (size_t)blockIdx.x * blockDim.x + threadIdx.x;
  size_t total = (size_t)R * C;
  if (idx >= total) return;
  int r = (int)(idx % (size_t)R);
  int c = (int)(idx / (size_t)R);
  dst[idx] = (bf16_t)src[(size_t)r * C + c];   // dst[C][R] K-major bf16
}

__global__ void gru_embed_bf16(const int* __restrict__ tokens,
                               const float* __restrict__ emb,
                               bf16_t* __restrict__ xb) {
  size_t idx = (size_t)blockIdx.x * blockDim.x + threadIdx.x;
  if (idx >= (size_t)BATCH * SEQ * EMBED) return;
  int e = (int)(idx % EMBED);
  int bs = (int)(idx / EMBED);
  int t = tokens[bs];
  xb[idx] = (bf16_t)emb[(size_t)t * EMBED + e];
}

__global__ void gru_zero_state(float* __restrict__ hf, bf16_t* __restrict__ hb,
                               unsigned* __restrict__ ctr) {
  int idx = blockIdx.x * blockDim.x + threadIdx.x;
  if (idx < BATCH * HID) {
    hf[idx] = 0.0f;
    hb[idx] = (bf16_t)0.0f;
  }
  if (idx == 0) *ctr = 0u;   // grid-barrier counter (monotonic per call)
}

// ------------------------- persistent GRU kernel ---------------------------

__device__ __forceinline__ void grid_barrier(unsigned* ctr, unsigned target) {
  __threadfence();            // release: flush this thread's h writes
  __syncthreads();
  if ((threadIdx.x & 31) == 0) {
    if (threadIdx.x == 0)
      __hip_atomic_fetch_add(ctr, 1u, __ATOMIC_RELEASE,
                             __HIP_MEMORY_SCOPE_AGENT);
    // lane 0 of every wave acquires, so each wave's caches are made coherent
    while (__hip_atomic_load(ctr, __ATOMIC_ACQUIRE,
                             __HIP_MEMORY_SCOPE_AGENT) < target) {
      __builtin_amdgcn_s_sleep(1);
    }
  }
  __syncthreads();
}

__global__ __launch_bounds__(128, 1) void gru_persistent(
    const bf16_t* __restrict__ UkT,    // [3072][1024] K-major bf16
    const bf16_t* __restrict__ WkT,    // [3072][256]  K-major bf16
    const bf16_t* __restrict__ xb,     // [64][512][256] bf16
    const int* __restrict__ tokens,    // [64][512]
    const float* __restrict__ bias,    // [2][3072]
    float* __restrict__ hf0, bf16_t* __restrict__ hb0,
    float* __restrict__ hf1, bf16_t* __restrict__ hb1,
    unsigned* __restrict__ barrier_ctr) {
  extern __shared__ bf16_t smem[];
  bf16_t* ldsUk = smem;                  // [3*16][UK_ROWPAD]
  bf16_t* ldsWk = smem + LDS_UK_ELEMS;   // [3*16][WK_ROWPAD]

  const int lane = threadIdx.x & 31;
  const int wave = threadIdx.x >> 5;
  const int half = lane >> 4;
  const int lm   = lane & 15;
  const int m0   = wave * 16;           // this wave's M tile
  const int nh0  = blockIdx.x * 16;     // this WG's H-column tile
  const int arow = m0 + lm;             // A-fragment row for this lane
  const int col  = nh0 + lm;            // B-fragment column for this lane

  // ---- stage step-invariant weight slices into LDS (once) ----
#if HAVE_TDM
  if (wave == 0) {
    for (int row = 0; row < 48; ++row) {       // row = g*16 + lc
      int g = row >> 4, lc = row & 15;
      tdm_load_row((unsigned)(row * UK_ROWPAD * 2),
                   UkT + ((size_t)(g * HID + nh0 + lc)) * HID, HID / 4);
      tdm_load_row((unsigned)((LDS_UK_ELEMS + row * WK_ROWPAD) * 2),
                   WkT + ((size_t)(g * HID + nh0 + lc)) * EMBED, EMBED / 4);
    }
    __builtin_amdgcn_s_wait_tensorcnt(0);
  }
#else
  for (int i = threadIdx.x; i < 3 * 16 * (HID / 8); i += 128) {
    int row = i / (HID / 8);
    int c8  = i % (HID / 8);
    int g = row >> 4, lc = row & 15;
    *(v8bf*)(ldsUk + row * UK_ROWPAD + c8 * 8) =
        *(const v8bf*)(UkT + ((size_t)(g * HID + nh0 + lc)) * HID + c8 * 8);
  }
  for (int i = threadIdx.x; i < 3 * 16 * (EMBED / 8); i += 128) {
    int row = i / (EMBED / 8);
    int c8  = i % (EMBED / 8);
    int g = row >> 4, lc = row & 15;
    *(v8bf*)(ldsWk + row * WK_ROWPAD + c8 * 8) =
        *(const v8bf*)(WkT + ((size_t)(g * HID + nh0 + lc)) * EMBED + c8 * 8);
  }
#endif
  __syncthreads();

  // LDS B-fragment base pointers (16B aligned: 2064*row + 32*half)
  const bf16_t* uz = ldsUk + (0 * 16 + lm) * UK_ROWPAD + half * 16;
  const bf16_t* ur = ldsUk + (1 * 16 + lm) * UK_ROWPAD + half * 16;
  const bf16_t* uh = ldsUk + (2 * 16 + lm) * UK_ROWPAD + half * 16;
  const bf16_t* wz = ldsWk + (0 * 16 + lm) * WK_ROWPAD + half * 16;
  const bf16_t* wr = ldsWk + (1 * 16 + lm) * WK_ROWPAD + half * 16;
  const bf16_t* wh = ldsWk + (2 * 16 + lm) * WK_ROWPAD + half * 16;

  // hoisted biases (b[0]=input, b[1]=recurrent; gate order z,r,h)
  const float b0z = bias[col],           b1z = bias[B3 + col];
  const float b0r = bias[HID + col],     b1r = bias[B3 + HID + col];
  const float b0h = bias[2 * HID + col], b1h = bias[B3 + 2 * HID + col];

  const bf16_t* aH0 = hb0 + (size_t)arow * HID + half * 16;  // A bases
  const bf16_t* aH1 = hb1 + (size_t)arow * HID + half * 16;
  const bf16_t* aX  = xb + (size_t)arow * SEQ * EMBED + half * 16;

  for (int s = 0; s < SEQ; ++s) {
    const bf16_t* ah  = (s & 1) ? aH1 : aH0;
    const float*  hfi = (s & 1) ? hf1 : hf0;
    float*        hfo = (s & 1) ? hf0 : hf1;
    bf16_t*       hbo = (s & 1) ? hb0 : hb1;

    v8f a_rz = {}, a_rr = {}, a_rh = {};
    v8f a_xz = {}, a_xr = {}, a_xh = {};

    // recurrent part: A from global h (bf16), B from LDS
#pragma unroll 4
    for (int k0 = 0; k0 < HID; k0 += 32) {
      v16bf a  = ld_a_frag(ah + k0);
      v16bf Bz = *(const v16bf*)(uz + k0);
      v16bf Br = *(const v16bf*)(ur + k0);
      v16bf Bh = *(const v16bf*)(uh + k0);
      a_rz = wmma_bf16(a, Bz, a_rz);
      a_rr = wmma_bf16(a, Br, a_rr);
      a_rh = wmma_bf16(a, Bh, a_rh);
    }
    // input part: A from precomputed x_bf16, B from LDS
    const bf16_t* ax = aX + (size_t)s * EMBED;
#pragma unroll
    for (int k0 = 0; k0 < EMBED; k0 += 32) {
      v16bf a  = ld_a_frag(ax + k0);
      v16bf Bz = *(const v16bf*)(wz + k0);
      v16bf Br = *(const v16bf*)(wr + k0);
      v16bf Bh = *(const v16bf*)(wh + k0);
      a_xz = wmma_bf16(a, Bz, a_xz);
      a_xr = wmma_bf16(a, Br, a_xr);
      a_xh = wmma_bf16(a, Bh, a_xh);
    }

    // fused GRU gates (reset_after=True) + mask, double-buffered state
#pragma unroll
    for (int v = 0; v < 8; ++v) {
      int m = m0 + half * 8 + v;   // C/D layout: lanes 0-15 M=v, 16-31 M=8+v
      float hold = hfi[(size_t)m * HID + col];
      int tok = tokens[(size_t)m * SEQ + s];
      float z  = sigm(a_xz[v] + a_rz[v] + b0z + b1z);
      float r  = sigm(a_xr[v] + a_rr[v] + b0r + b1r);
      float hh = tanh_fast(a_xh[v] + b0h + r * (a_rh[v] + b1h));
      float hn = z * hold + (1.0f - z) * hh;
      if (tok == 0) hn = hold;     // mask_zero: carry state
      hfo[(size_t)m * HID + col] = hn;
      hbo[(size_t)m * HID + col] = (bf16_t)hn;
    }

    grid_barrier(barrier_ctr, (unsigned)NWG * (unsigned)(s + 1));
  }
}

// ------------------------- output projection (fp32 WMMA) -------------------
__global__ __launch_bounds__(128) void gru_out_proj(
    const float* __restrict__ hf, const float* __restrict__ Wout,
    float* __restrict__ out) {
  const int lane = threadIdx.x & 31;
  const int wave = threadIdx.x >> 5;
  const int half = lane >> 4;
  const int lm   = lane & 15;
  const int m0   = wave * 16;
  const int n0   = blockIdx.x * 16;

  const float* ap = hf + (size_t)(m0 + lm) * HID + half * 2;
  const float* bp = Wout + (n0 + lm);
  v8f acc = {};
#pragma unroll 8
  for (int k0 = 0; k0 < HID; k0 += 4) {
    v2f a = *(const v2f*)(ap + k0);
    v2f b;
    b.x = bp[(size_t)(k0 + half * 2 + 0) * VOCAB];
    b.y = bp[(size_t)(k0 + half * 2 + 1) * VOCAB];
    acc = __builtin_amdgcn_wmma_f32_16x16x4_f32(
        false, a, false, b, (short)0, acc, false, false);
  }
#pragma unroll
  for (int v = 0; v < 8; ++v) {
    int m = m0 + half * 8 + v;
    out[(size_t)m * VOCAB + n0 + lm] = acc[v];
  }
}

// ---------------------------------------------------------------------------

extern "C" void kernel_launch(void* const* d_in, const int* in_sizes, int n_in,
                              void* d_out, int out_size, void* d_ws,
                              size_t ws_size, hipStream_t stream) {
  const int*   tokens = (const int*)d_in[0];
  const float* emb    = (const float*)d_in[1];
  const float* Wk     = (const float*)d_in[2];
  const float* Uk     = (const float*)d_in[3];
  const float* bias   = (const float*)d_in[4];
  const float* Wout   = (const float*)d_in[5];
  float* out = (float*)d_out;

  // workspace layout (bytes)
  char* ws = (char*)d_ws;
  bf16_t*   UkT = (bf16_t*)(ws + 0);          //  6,291,456  [3072][1024]
  bf16_t*   WkT = (bf16_t*)(ws + 6291456);    //  1,572,864  [3072][256]
  bf16_t*   xb  = (bf16_t*)(ws + 7864320);    // 16,777,216  [64][512][256]
  float*    hf0 = (float*)(ws + 24641536);    //    262,144
  float*    hf1 = (float*)(ws + 24903680);    //    262,144
  bf16_t*   hb0 = (bf16_t*)(ws + 25165824);   //    131,072
  bf16_t*   hb1 = (bf16_t*)(ws + 25296896);   //    131,072
  unsigned* ctr = (unsigned*)(ws + 25427968); //        256  (~25.4 MB total)

  // allow >64KB dynamic LDS for the persistent kernel (capture-safe host call)
  static_assert(LDS_BYTES <= 160 * 1024, "LDS budget");
  hipFuncSetAttribute((const void*)gru_persistent,
                      hipFuncAttributeMaxDynamicSharedMemorySize, LDS_BYTES);

  gru_transpose_bf16<<<(HID * B3 + 255) / 256, 256, 0, stream>>>(Uk, UkT, HID, B3);
  gru_transpose_bf16<<<(EMBED * B3 + 255) / 256, 256, 0, stream>>>(Wk, WkT, EMBED, B3);
  gru_embed_bf16<<<(BATCH * SEQ * EMBED + 255) / 256, 256, 0, stream>>>(tokens, emb, xb);
  gru_zero_state<<<(BATCH * HID + 255) / 256, 256, 0, stream>>>(hf0, hb0, ctr);

  // all 512 timesteps in ONE persistent launch (64 co-resident WGs)
  gru_persistent<<<dim3(NWG), dim3(128), LDS_BYTES, stream>>>(
      UkT, WkT, xb, tokens, bias, hf0, hb0, hf1, hb1, ctr);
  // s=511 (odd) wrote hf0 -> final state

  gru_out_proj<<<dim3(VOCAB / 16), dim3(128), 0, stream>>>(hf0, Wout, out);
}
